// OppoModelNet_88441966559674
// MI455X (gfx1250) — matrix operand order repaired and checked by reading, not compiled, not run on hardware
//
#include <hip/hip_runtime.h>
#include <hip/hip_bf16.h>
#include <cstddef>
#include <cstdint>

// ---------------------------------------------------------------------------
// MI455X / gfx1250: wave32, WMMA f32 = A(16x32 f16) x B(32x16 f16) + C(16x16 f32)
// ---------------------------------------------------------------------------

typedef __attribute__((ext_vector_type(16))) _Float16 v16h;
typedef __attribute__((ext_vector_type(8)))  float    v8f;

#define NNODES 65536
#define NEDGES 524288

union F16x16 {
  v16h v;
  uint4 q[2];
};

__device__ __forceinline__ v8f wmma16(v16h a, v16h b, v8f c) {
  return __builtin_amdgcn_wmma_f32_16x16x32_f16(false, a, false, b, (short)0, c,
                                                false, false);
}

__device__ __forceinline__ v8f zero8() {
  v8f z;
#pragma unroll
  for (int i = 0; i < 8; ++i) z[i] = 0.0f;
  return z;
}

__device__ __forceinline__ float sigmoidf_(float x) {
  return 1.0f / (1.0f + __expf(-x));
}

// A fragment (16x32, f16). Per ISA table: lanes 0-15 hold row M=lane with
// K = {kc+0..7, kc+16..23}; lanes 16-31 hold row M=lane-16 with
// K = {kc+8..15, kc+24..31}. rowp already points at the lane's row.
__device__ __forceinline__ v16h a_frag_f16(const _Float16* rowp, int kc) {
  const int lane = threadIdx.x & 31;
  const int kb = (lane < 16) ? 0 : 8;
  const _Float16* p = rowp + kc + kb;
  F16x16 f;
  f.q[0] = *reinterpret_cast<const uint4*>(p);
  f.q[1] = *reinterpret_cast<const uint4*>(p + 16);
  return f.v;
}

// Same fragment but source rows are f32; convert on the fly (b128 loads + cvt).
__device__ __forceinline__ v16h a_frag_f32(const float* rowp, int kc) {
  const int lane = threadIdx.x & 31;
  const int kb = (lane < 16) ? 0 : 8;
  const float* p = rowp + kc + kb;
  F16x16 f;
#pragma unroll
  for (int i = 0; i < 8; ++i) f.v[i] = (_Float16)p[i];
#pragma unroll
  for (int i = 0; i < 8; ++i) f.v[8 + i] = (_Float16)p[16 + i];
  return f.v;
}

// B fragment (32x16, f16) from row-major weight W[Nout][ldk]; B[k][n] = W[n][k].
// Lanes 0-15: column n0+lane, K = {kc+0..7, kc+16..23}; lanes 16-31: column
// n0+lane-16, K = {kc+8..15, kc+24..31}. Two b128 loads per lane.
__device__ __forceinline__ v16h b_frag(const _Float16* W, int ldk, int n0, int kc) {
  const int lane = threadIdx.x & 31;
  const _Float16* p =
      W + (size_t)(n0 + (lane & 15)) * ldk + kc + ((lane < 16) ? 0 : 8);
  F16x16 f;
  f.q[0] = *reinterpret_cast<const uint4*>(p);
  f.q[1] = *reinterpret_cast<const uint4*>(p + 16);
  return f.v;
}

// ---------------------------------------------------------------------------
// Utility kernels
// ---------------------------------------------------------------------------
__global__ void cvt_f32_to_f16(const float* __restrict__ in,
                               _Float16* __restrict__ out, int n) {
  int i = blockIdx.x * blockDim.x + threadIdx.x;
  if (i < n) out[i] = (_Float16)in[i];
}

__global__ void zero_f32(float* __restrict__ p, int n) {
  int i = blockIdx.x * blockDim.x + threadIdx.x;
  if (i < n) p[i] = 0.0f;
}

// ---------------------------------------------------------------------------
// Kernel 1: node encoder, fully fused per 16-node tile:
//   t1 = relu(obs @ w1a^T + b1a)             [16 x 512]  (LDS)
//   x  = t1 @ w1b^T + b1b                    [16 x 256]  (LDS)
//   gates = x @ w_ih^T + h0 @ w_hh^T + b     [16 x 1024] (registers)
//   c1 = sig(f)*c0 + sig(i)*tanh(g); h1 = sig(o)*tanh(c1)   -> d_out (f32)
//   nf = relu(h1) @ w1^T + b1                [16 x 128]  -> scratch (f16)
// 8 waves/WG; LDS rows padded +8 f16 to spread banks.
// ---------------------------------------------------------------------------
__global__ __launch_bounds__(256) void encoder_kernel(
    const float* __restrict__ obs, const float* __restrict__ h0,
    const float* __restrict__ c0,
    const _Float16* __restrict__ w1a, const float* __restrict__ b1a,
    const _Float16* __restrict__ w1b, const float* __restrict__ b1b,
    const _Float16* __restrict__ w_ih, const _Float16* __restrict__ w_hh,
    const float* __restrict__ b_ih, const float* __restrict__ b_hh,
    const _Float16* __restrict__ w1, const float* __restrict__ b1,
    float* __restrict__ h1_out, float* __restrict__ c1_out,
    _Float16* __restrict__ nf_out) {
  __shared__ _Float16 ldsA[16 * 520];  // t1 (stride 520), later relu(h1) (stride 264)
  __shared__ _Float16 ldsB[16 * 264];  // x (stride 264)

  const int m0 = blockIdx.x * 16;
  const int wave = threadIdx.x >> 5;
  const int lane = threadIdx.x & 31;
  const int lrow = lane & 15;
  const int mb = (lane < 16) ? 0 : 8;

  // ---- Stage 1: t1 = relu(obs @ w1a^T + b1a): 32 tiles, 4 per wave
  {
    const float* arow = obs + (size_t)(m0 + lrow) * 256;
    v8f acc[4];
#pragma unroll
    for (int t = 0; t < 4; ++t) acc[t] = zero8();
    for (int kc = 0; kc < 256; kc += 32) {
      v16h a = a_frag_f32(arow, kc);
#pragma unroll
      for (int t = 0; t < 4; ++t)
        acc[t] = wmma16(a, b_frag(w1a, 256, (wave * 4 + t) * 16, kc), acc[t]);
    }
#pragma unroll
    for (int t = 0; t < 4; ++t) {
      int n = (wave * 4 + t) * 16 + lrow;
      float bias = b1a[n];
#pragma unroll
      for (int v = 0; v < 8; ++v)
        ldsA[(mb + v) * 520 + n] = (_Float16)fmaxf(acc[t][v] + bias, 0.0f);
    }
  }
  __syncthreads();

  // ---- Stage 2: x = t1 @ w1b^T + b1b: 16 tiles, 2 per wave, K=512
  {
    const _Float16* arow = ldsA + lrow * 520;
    v8f acc[2];
    acc[0] = zero8();
    acc[1] = zero8();
    for (int kc = 0; kc < 512; kc += 32) {
      v16h a = a_frag_f16(arow, kc);
#pragma unroll
      for (int t = 0; t < 2; ++t)
        acc[t] = wmma16(a, b_frag(w1b, 512, (wave * 2 + t) * 16, kc), acc[t]);
    }
#pragma unroll
    for (int t = 0; t < 2; ++t) {
      int n = (wave * 2 + t) * 16 + lrow;
      float bias = b1b[n];
#pragma unroll
      for (int v = 0; v < 8; ++v)
        ldsB[(mb + v) * 264 + n] = (_Float16)(acc[t][v] + bias);
    }
  }
  __syncthreads();

  // ---- Stage 3: LSTM. Wave owns gate columns [32*wave, 32*wave+32) so that
  // i/f/g/o accumulators for the same (m,n) live in the same lane/VGPR slot.
  {
    const _Float16* xrow = ldsB + lrow * 264;
    const float* hrow = h0 + (size_t)(m0 + lrow) * 256;
    v8f acc[4][2];
#pragma unroll
    for (int g = 0; g < 4; ++g) {
      acc[g][0] = zero8();
      acc[g][1] = zero8();
    }
    for (int kc = 0; kc < 256; kc += 32) {
      v16h ax = a_frag_f16(xrow, kc);
      v16h ah = a_frag_f32(hrow, kc);
#pragma unroll
      for (int g = 0; g < 4; ++g) {
#pragma unroll
        for (int p = 0; p < 2; ++p) {
          int n0 = g * 256 + wave * 32 + p * 16;
          acc[g][p] = wmma16(ax, b_frag(w_ih, 256, n0, kc), acc[g][p]);
          acc[g][p] = wmma16(ah, b_frag(w_hh, 256, n0, kc), acc[g][p]);
        }
      }
    }
#pragma unroll
    for (int p = 0; p < 2; ++p) {
      int n = wave * 32 + p * 16 + lrow;
      float bi = b_ih[n] + b_hh[n];
      float bf = b_ih[256 + n] + b_hh[256 + n];
      float bg = b_ih[512 + n] + b_hh[512 + n];
      float bo = b_ih[768 + n] + b_hh[768 + n];
#pragma unroll
      for (int v = 0; v < 8; ++v) {
        size_t idx = (size_t)(m0 + mb + v) * 256 + n;
        float iv = sigmoidf_(acc[0][p][v] + bi);
        float fv = sigmoidf_(acc[1][p][v] + bf);
        float gv = tanhf(acc[2][p][v] + bg);
        float ov = sigmoidf_(acc[3][p][v] + bo);
        float c1v = fv * c0[idx] + iv * gv;
        float h1v = ov * tanhf(c1v);
        c1_out[idx] = c1v;
        h1_out[idx] = h1v;
        ldsA[(mb + v) * 264 + n] = (_Float16)fmaxf(h1v, 0.0f);
      }
    }
  }
  __syncthreads();

  // ---- Stage 4: nf = relu(h1) @ w1^T + b1: 8 tiles, 1 per wave
  {
    const _Float16* arow = ldsA + lrow * 264;
    v8f acc = zero8();
    for (int kc = 0; kc < 256; kc += 32)
      acc = wmma16(a_frag_f16(arow, kc), b_frag(w1, 256, wave * 16, kc), acc);
    int n = wave * 16 + lrow;
    float bias = b1[n];
#pragma unroll
    for (int v = 0; v < 8; ++v)
      nf_out[(size_t)(m0 + mb + v) * 128 + n] = (_Float16)(acc[v] + bias);
  }
}

// ---------------------------------------------------------------------------
// Kernel 2: edge MLP + fused segment-sum, per 16-edge tile:
//   h  = relu([nf[src] ++ nf[dst]] @ we^T + be)   [16 x 256] (LDS)
//   ef = h @ we2^T + be2                          [16 x 128]
//   agg[dst] += ef   via global_atomic_add_f32
// Gather is fused into the A-fragment loader (two row pointers, K-chunk select).
// ---------------------------------------------------------------------------
__global__ __launch_bounds__(256) void edge_kernel(
    const _Float16* __restrict__ nf, const int* __restrict__ src,
    const int* __restrict__ dst,
    const _Float16* __restrict__ we, const float* __restrict__ be,
    const _Float16* __restrict__ we2, const float* __restrict__ be2,
    float* __restrict__ agg) {
  __shared__ _Float16 lds[16 * 264];

  const int e0 = blockIdx.x * 16;
  const int wave = threadIdx.x >> 5;
  const int lane = threadIdx.x & 31;
  const int lrow = lane & 15;
  const int mb = (lane < 16) ? 0 : 8;

  const _Float16* srow = nf + (size_t)src[e0 + lrow] * 128;
  const _Float16* drow = nf + (size_t)dst[e0 + lrow] * 128 - 128;  // +kc lands right

  // ---- Stage 1
  {
    v8f acc[2];
    acc[0] = zero8();
    acc[1] = zero8();
    for (int kc = 0; kc < 256; kc += 32) {
      const _Float16* arow = (kc < 128) ? srow : drow;  // uniform branch
      v16h a = a_frag_f16(arow, kc);
#pragma unroll
      for (int t = 0; t < 2; ++t)
        acc[t] = wmma16(a, b_frag(we, 256, (wave * 2 + t) * 16, kc), acc[t]);
    }
#pragma unroll
    for (int t = 0; t < 2; ++t) {
      int n = (wave * 2 + t) * 16 + lrow;
      float bias = be[n];
#pragma unroll
      for (int v = 0; v < 8; ++v)
        lds[(mb + v) * 264 + n] = (_Float16)fmaxf(acc[t][v] + bias, 0.0f);
    }
  }
  __syncthreads();

  // ---- Stage 2 + scatter-add
  {
    const _Float16* arow = lds + lrow * 264;
    v8f acc = zero8();
    for (int kc = 0; kc < 256; kc += 32)
      acc = wmma16(a_frag_f16(arow, kc), b_frag(we2, 256, wave * 16, kc), acc);
    int n = wave * 16 + lrow;
    float bias = be2[n];
#pragma unroll
    for (int v = 0; v < 8; ++v) {
      int d = dst[e0 + mb + v];
      atomicAdd(&agg[(size_t)d * 128 + n], acc[v] + bias);
    }
  }
}

// ---------------------------------------------------------------------------
// Kernel 3: node MLP + readout, per 16-node tile:
//   nh  = relu([nf ++ agg] @ wn^T + bn)  [16 x 256] (LDS)
//   nf2 = nh @ wn2^T + bn2               [16 x 128] (LDS)
//   rh  = relu(nf2 @ wr^T + br)          [16 x 256] (LDS)
//   logits = rh @ wr2^T + br2            [16 x 8]   (scalar stage, tiny)
// ---------------------------------------------------------------------------
__global__ __launch_bounds__(256) void node_readout_kernel(
    const _Float16* __restrict__ nf, const _Float16* __restrict__ aggh,
    const _Float16* __restrict__ wn, const float* __restrict__ bn,
    const _Float16* __restrict__ wn2, const float* __restrict__ bn2,
    const _Float16* __restrict__ wr, const float* __restrict__ br,
    const float* __restrict__ wr2, const float* __restrict__ br2,
    float* __restrict__ logits) {
  __shared__ _Float16 ldsA[16 * 264];  // nh, later rh
  __shared__ _Float16 ldsB[16 * 136];  // nf2

  const int m0 = blockIdx.x * 16;
  const int wave = threadIdx.x >> 5;
  const int lane = threadIdx.x & 31;
  const int lrow = lane & 15;
  const int mb = (lane < 16) ? 0 : 8;

  const _Float16* r0 = nf + (size_t)(m0 + lrow) * 128;
  const _Float16* r1 = aggh + (size_t)(m0 + lrow) * 128 - 128;

  // ---- Stage 1: nh
  {
    v8f acc[2];
    acc[0] = zero8();
    acc[1] = zero8();
    for (int kc = 0; kc < 256; kc += 32) {
      const _Float16* arow = (kc < 128) ? r0 : r1;
      v16h a = a_frag_f16(arow, kc);
#pragma unroll
      for (int t = 0; t < 2; ++t)
        acc[t] = wmma16(a, b_frag(wn, 256, (wave * 2 + t) * 16, kc), acc[t]);
    }
#pragma unroll
    for (int t = 0; t < 2; ++t) {
      int n = (wave * 2 + t) * 16 + lrow;
      float bias = bn[n];
#pragma unroll
      for (int v = 0; v < 8; ++v)
        ldsA[(mb + v) * 264 + n] = (_Float16)fmaxf(acc[t][v] + bias, 0.0f);
    }
  }
  __syncthreads();

  // ---- Stage 2: nf2
  {
    const _Float16* arow = ldsA + lrow * 264;
    v8f acc = zero8();
    for (int kc = 0; kc < 256; kc += 32)
      acc = wmma16(a_frag_f16(arow, kc), b_frag(wn2, 256, wave * 16, kc), acc);
    int n = wave * 16 + lrow;
    float bias = bn2[n];
#pragma unroll
    for (int v = 0; v < 8; ++v)
      ldsB[(mb + v) * 136 + n] = (_Float16)(acc[v] + bias);
  }
  __syncthreads();

  // ---- Stage 3: rh (K = 128)
  {
    const _Float16* arow = ldsB + lrow * 136;
    v8f acc[2];
    acc[0] = zero8();
    acc[1] = zero8();
    for (int kc = 0; kc < 128; kc += 32) {
      v16h a = a_frag_f16(arow, kc);
#pragma unroll
      for (int t = 0; t < 2; ++t)
        acc[t] = wmma16(a, b_frag(wr, 128, (wave * 2 + t) * 16, kc), acc[t]);
    }
#pragma unroll
    for (int t = 0; t < 2; ++t) {
      int n = (wave * 2 + t) * 16 + lrow;
      float bias = br[n];
#pragma unroll
      for (int v = 0; v < 8; ++v)
        ldsA[(mb + v) * 264 + n] = (_Float16)fmaxf(acc[t][v] + bias, 0.0f);
    }
  }
  __syncthreads();

  // ---- Stage 4: logits, 8 outputs per node — tiny, do it scalar from LDS
  if (threadIdx.x < 128) {
    int m = threadIdx.x >> 3;
    int a = threadIdx.x & 7;
    const _Float16* r = ldsA + m * 264;
    const float* w = wr2 + a * 256;
    float s = br2[a];
#pragma unroll 8
    for (int k = 0; k < 256; ++k) s += (float)r[k] * w[k];
    logits[(size_t)(m0 + m) * 8 + a] = s;
  }
}

// ---------------------------------------------------------------------------
// Host launcher
// ---------------------------------------------------------------------------
extern "C" void kernel_launch(void* const* d_in, const int* in_sizes, int n_in,
                              void* d_out, int out_size, void* d_ws,
                              size_t ws_size, hipStream_t stream) {
  (void)in_sizes; (void)n_in; (void)out_size; (void)ws_size;

  const float* obs = (const float*)d_in[0];
  const float* h0 = (const float*)d_in[1];
  const float* c0 = (const float*)d_in[2];
  const int* src = (const int*)d_in[3];
  const int* dst = (const int*)d_in[4];
  const float* w1a_f = (const float*)d_in[5];
  const float* b1a = (const float*)d_in[6];
  const float* w1b_f = (const float*)d_in[7];
  const float* b1b = (const float*)d_in[8];
  const float* wih_f = (const float*)d_in[9];
  const float* bih = (const float*)d_in[10];
  const float* whh_f = (const float*)d_in[11];
  const float* bhh = (const float*)d_in[12];
  const float* w1_f = (const float*)d_in[13];
  const float* b1 = (const float*)d_in[14];
  const float* we_f = (const float*)d_in[15];
  const float* be = (const float*)d_in[16];
  const float* we2_f = (const float*)d_in[17];
  const float* be2 = (const float*)d_in[18];
  const float* wn_f = (const float*)d_in[19];
  const float* bn = (const float*)d_in[20];
  const float* wn2_f = (const float*)d_in[21];
  const float* bn2 = (const float*)d_in[22];
  const float* wr_f = (const float*)d_in[23];
  const float* br = (const float*)d_in[24];
  const float* wr2 = (const float*)d_in[25];
  const float* br2 = (const float*)d_in[26];

  // Workspace layout (66 MB total):
  //   [0,16MB)   nf   f16 [N,128]
  //   [16,48MB)  agg  f32 [N,128]
  //   [48,64MB)  aggh f16 [N,128]
  //   [64,66MB)  f16 weights (exactly 1M elements)
  char* ws = (char*)d_ws;
  _Float16* nf = (_Float16*)ws;
  float* agg = (float*)(ws + (size_t)16 * 1024 * 1024);
  _Float16* aggh = (_Float16*)(ws + (size_t)48 * 1024 * 1024);
  _Float16* wp = (_Float16*)(ws + (size_t)64 * 1024 * 1024);
  _Float16* w1a = wp; wp += 512 * 256;
  _Float16* w1b = wp; wp += 256 * 512;
  _Float16* wih = wp; wp += 1024 * 256;
  _Float16* whh = wp; wp += 1024 * 256;
  _Float16* w1 = wp;  wp += 128 * 256;
  _Float16* we = wp;  wp += 256 * 256;
  _Float16* we2 = wp; wp += 128 * 256;
  _Float16* wn = wp;  wp += 256 * 256;
  _Float16* wn2 = wp; wp += 128 * 256;
  _Float16* wr = wp;  wp += 256 * 128;

  auto cvt = [&](const float* s, _Float16* d, int n) {
    cvt_f32_to_f16<<<dim3((n + 255) / 256), dim3(256), 0, stream>>>(s, d, n);
  };
  cvt(w1a_f, w1a, 512 * 256);
  cvt(w1b_f, w1b, 256 * 512);
  cvt(wih_f, wih, 1024 * 256);
  cvt(whh_f, whh, 1024 * 256);
  cvt(w1_f, w1, 128 * 256);
  cvt(we_f, we, 256 * 256);
  cvt(we2_f, we2, 128 * 256);
  cvt(wn_f, wn, 256 * 256);
  cvt(wn2_f, wn2, 128 * 256);
  cvt(wr_f, wr, 256 * 128);

  zero_f32<<<dim3(NNODES * 128 / 1024), dim3(1024), 0, stream>>>(agg,
                                                                 NNODES * 128);

  float* out = (float*)d_out;
  float* logits = out;                           // [N, 8]
  float* h1 = out + (size_t)NNODES * 8;          // [N, 256]
  float* c1 = h1 + (size_t)NNODES * 256;         // [N, 256]

  encoder_kernel<<<dim3(NNODES / 16), dim3(256), 0, stream>>>(
      obs, h0, c0, w1a, b1a, w1b, b1b, wih, whh, bih, bhh, w1, b1, h1, c1, nf);

  edge_kernel<<<dim3(NEDGES / 16), dim3(256), 0, stream>>>(nf, src, dst, we, be,
                                                           we2, be2, agg);

  cvt((const float*)agg, aggh, NNODES * 128);

  node_readout_kernel<<<dim3(NNODES / 16), dim3(256), 0, stream>>>(
      nf, aggh, wn, bn, wn2, bn2, wr, br, wr2, br2, logits);
}
